// Model_87256555585715
// MI455X (gfx1250) — matrix operand order repaired and checked by reading, not compiled, not run on hardware
//
#include <hip/hip_runtime.h>
#include <hip/hip_bf16.h>
#include <stdint.h>

// ---------------------------------------------------------------------------
// Types
// ---------------------------------------------------------------------------
typedef unsigned short u16;
typedef __attribute__((ext_vector_type(16))) __bf16        v16bf;
typedef __attribute__((ext_vector_type(8)))  float         v8f;
typedef __attribute__((ext_vector_type(4)))  unsigned int  uv4;
typedef __attribute__((ext_vector_type(4)))  unsigned int  tdm_v4u;
typedef __attribute__((ext_vector_type(8)))  int           tdm_v8i;
typedef __attribute__((ext_vector_type(4)))  int           tdm_v4i;

union Frag   { v16bf v; uv4 q[2]; };
union Pack16 { u16 h[16]; uv4 q[2]; };
union Pack8  { u16 h[8];  uv4 q; };

__device__ __forceinline__ u16 f2bf(float f) {
  union { __hip_bfloat16 b; u16 u; } cv; cv.b = __float2bfloat16(f); return cv.u;
}
__device__ __forceinline__ float bf2f(u16 u) {
  union { u16 u; __hip_bfloat16 b; } cv; cv.u = u; return __bfloat162float(cv.b);
}

// ---------------------------------------------------------------------------
// Problem constants
// ---------------------------------------------------------------------------
static constexpr int CHN  = 1000;   // channels
static constexpr int CHP  = 1024;   // padded channels (WMMA M-tiles)
static constexpr int SEQ  = 1024;   // input sequence length (= GEMM K)
static constexpr int PRED = 192;    // prediction length (= GEMM M)
static constexpr int NB   = 32;     // batch
static constexpr int KD   = 16384;  // k-means feature dim = NB * 512
static constexpr int KMI  = 50;     // k-means iterations
static constexpr int KBLK = 128;    // head GEMM K-slab staged per TDM

// Workspace layout (bytes, all 256-aligned)
static constexpr size_t OFF_MEAN = 0;                       // 32000 f32
static constexpr size_t OFF_RSTD = 128000;                  // 32000 f32
static constexpr size_t OFF_STD  = 256000;                  // 32000 f32
static constexpr size_t OFF_ZZ   = 384000;                  // 1024 f32
static constexpr size_t OFF_CF   = 388096;                  // 3*16384 f32
static constexpr size_t OFF_CB   = 584704;                  // 16*16384 bf16 (row 3 zero)
static constexpr size_t OFF_PCC  = 1108992;                 // 64*3 f32 partial |c|^2
static constexpr size_t OFF_LAB  = 1110016;                 // 1024 i32
static constexpr size_t OFF_BIAS = 1114112;                 // 3*192 f32
static constexpr size_t OFF_WF   = 1116672;                 // 3*192*1024 bf16
static constexpr size_t OFF_ZB   = 2296320;                 // 1024*16384 bf16
static constexpr size_t OFF_XT   = 35850752;                // 1000*32*1024 bf16

// ---------------------------------------------------------------------------
// 1) Per-(b,ch) mean / rstd / std over L=1024
// ---------------------------------------------------------------------------
__global__ void stats_kernel(const float* __restrict__ x,
                             float* __restrict__ mean, float* __restrict__ rstd,
                             float* __restrict__ stdv) {
  int n = blockIdx.x * 256 + threadIdx.x;
  int b = blockIdx.y;
  if (n >= CHN) return;
  const float* p = x + (size_t)b * SEQ * CHN + n;
  float s = 0.f, sq = 0.f;
  for (int l = 0; l < SEQ; ++l) { float v = p[(size_t)l * CHN]; s += v; sq += v * v; }
  float m   = s * (1.f / SEQ);
  float var = (sq - s * s * (1.f / SEQ)) * (1.f / (SEQ - 1)) + 1e-5f;
  float rs  = rsqrtf(var);
  mean[b * CHN + n] = m;
  rstd[b * CHN + n] = rs;
  stdv[b * CHN + n] = var * rs; // sqrt(var)
}

// ---------------------------------------------------------------------------
// 2) Normalize + build bf16 operands:
//    xt[ch][b][s] = xn  (B operand for head GEMM, s contiguous)
//    zb[ch][b*512+l] = 0.5*(xn[2l]+xn[2l+1])  (k-means points, row-contiguous)
// ---------------------------------------------------------------------------
__global__ void prep_kernel(const float* __restrict__ x,
                            const float* __restrict__ mean, const float* __restrict__ rstd,
                            u16* __restrict__ xt, u16* __restrict__ zb) {
  __shared__ u16 tile[64][72]; // [s_local][n_local], padded
  int b  = blockIdx.z;
  int s0 = blockIdx.y * 64;
  int n0 = blockIdx.x * 64;
  int t  = threadIdx.x;
  // phase 1: coalesced load over n, normalize, stash bf16
  {
    int nl = t & 63, sr = t >> 6;
    int n = n0 + nl;
    float m = 0.f, rs = 0.f;
    if (n < CHN) { m = mean[b * CHN + n]; rs = rstd[b * CHN + n]; }
    for (int i = 0; i < 16; ++i) {
      int sl = sr + i * 4;
      float v = (n < CHN) ? x[(size_t)b * SEQ * CHN + (size_t)(s0 + sl) * CHN + n] : 0.f;
      tile[sl][nl] = f2bf((v - m) * rs);
    }
  }
  __syncthreads();
  int nl = t >> 2;
  int ch = n0 + nl;
  // phase 2a: write xt rows (contiguous in s)
  if (ch < CHN) {
    int sc = (t & 3) * 16;
    Pack16 pk;
    #pragma unroll
    for (int j = 0; j < 16; ++j) pk.h[j] = tile[sc + j][nl];
    uv4* dst = (uv4*)(xt + (size_t)ch * (NB * SEQ) + (size_t)b * SEQ + s0 + sc);
    dst[0] = pk.q[0]; dst[1] = pk.q[1];
  }
  // phase 2b: write z rows (x_L, contiguous in l); padded channels get zeros
  {
    int lc = (t & 3) * 8;
    Pack8 pk;
    #pragma unroll
    for (int j = 0; j < 8; ++j) {
      int l = lc + j;
      float z = 0.5f * (bf2f(tile[2 * l][nl]) + bf2f(tile[2 * l + 1][nl]));
      pk.h[j] = f2bf(z);
    }
    *(uv4*)(zb + (size_t)(n0 + nl) * KD + (size_t)b * 512 + (s0 >> 1) + lc) = pk.q;
  }
}

// ---------------------------------------------------------------------------
// 3) zz[ch] = |z_ch|^2
// ---------------------------------------------------------------------------
__global__ void zz_kernel(const u16* __restrict__ zb, float* __restrict__ zz) {
  __shared__ float red[256];
  int ch = blockIdx.x, t = threadIdx.x;
  const u16* row = zb + (size_t)ch * KD;
  float s = 0.f;
  for (int i = 0; i < KD / 256; ++i) { float v = bf2f(row[i * 256 + t]); s += v * v; }
  red[t] = s; __syncthreads();
  for (int k = 128; k > 0; k >>= 1) { if (t < k) red[t] += red[t + k]; __syncthreads(); }
  if (t == 0) zz[ch] = red[0];
}

// ---------------------------------------------------------------------------
// 4) Fuse weights: Wfull[k][p][s] = 0.5*(Wg[k,p,s/2] +/- Wu[p,s/2]); biasf=bg+bu
// ---------------------------------------------------------------------------
__global__ void wfuse_kernel(const float* __restrict__ Wg, const float* __restrict__ Wu,
                             const float* __restrict__ bg, const float* __restrict__ bu,
                             u16* __restrict__ wfull, float* __restrict__ biasf) {
  int idx = blockIdx.x * 256 + threadIdx.x;
  if (idx < 3 * PRED * SEQ) {
    int k = idx / (PRED * SEQ), r = idx % (PRED * SEQ);
    int p = r / SEQ, s = r % SEQ, l = s >> 1;
    float g = Wg[(size_t)k * PRED * 512 + (size_t)p * 512 + l];
    float u = Wu[(size_t)p * 512 + l];
    wfull[idx] = f2bf(0.5f * ((s & 1) ? (g - u) : (g + u)));
  }
  if (idx < 3 * PRED) biasf[idx] = bg[idx] + bu[idx % PRED];
}

// ---------------------------------------------------------------------------
// 5) K-means init: c0 = z[0..2]; |c|^2 partials seeded from zz
// ---------------------------------------------------------------------------
__global__ void kinit_kernel(const u16* __restrict__ zb, const float* __restrict__ zz,
                             u16* __restrict__ cB, float* __restrict__ cF,
                             float* __restrict__ pcc) {
  int idx = blockIdx.x * 256 + threadIdx.x;
  if (idx < 16 * KD) {
    int row = idx >> 14, d = idx & (KD - 1);
    if (row < 3) { u16 u = zb[(size_t)row * KD + d]; cB[idx] = u; cF[row * KD + d] = bf2f(u); }
    else cB[idx] = 0;
  }
  if (idx < 192) pcc[idx] = (idx < 3) ? zz[idx] : 0.f;
}

// ---------------------------------------------------------------------------
// 6) K-means assign: WMMA distance cross-term, argmin labels.
//    1 wave per block, 16 channels per block (64 blocks cover 1024 padded).
//    Centroid columns >= 3 alias the single zero row (broadcast, cache-hot).
// ---------------------------------------------------------------------------
__global__ void __launch_bounds__(32) assign_kernel(
    const u16* __restrict__ zb, const u16* __restrict__ cB,
    const float* __restrict__ zz, const float* __restrict__ pcc,
    int* __restrict__ labels) {
  __shared__ float ccs[3];
  __shared__ float dmat[16][4];
  int t = threadIdx.x;
  if (t < 3) { float s = 0.f; for (int j = 0; j < 64; ++j) s += pcc[j * 3 + t]; ccs[t] = s; }
  __syncthreads();
  int ch0 = blockIdx.x * 16;
  int lane = t & 31, hi = lane >> 4, m = lane & 15;
  const u16* zrow = zb + (size_t)(ch0 + m) * KD;            // A: M rows = channels
  const u16* crow = cB + (size_t)(m < 3 ? m : 3) * KD;      // B: cols>=3 -> shared zero row
  v8f acc = {0.f, 0.f, 0.f, 0.f, 0.f, 0.f, 0.f, 0.f};
  for (int k0 = 0; k0 < KD; k0 += 32) {
    Frag a, b;
    a.q[0] = *(const uv4*)(zrow + k0 + hi * 8);        // K 0..7 / 8..15
    a.q[1] = *(const uv4*)(zrow + k0 + 16 + hi * 8);   // K 16..23 / 24..31
    b.q[0] = *(const uv4*)(crow + k0 + hi * 8);        // K 0..7 / 8..15
    b.q[1] = *(const uv4*)(crow + k0 + 16 + hi * 8);   // K 16..23 / 24..31
    acc = __builtin_amdgcn_wmma_f32_16x16x32_bf16(false, a.v, false, b.v,
                                                  (short)0, acc, false, false);
  }
  if ((lane & 15) < 3) {
    #pragma unroll
    for (int r = 0; r < 8; ++r) dmat[r + hi * 8][lane & 15] = acc[r];
  }
  __syncthreads();
  if (t < 16) {
    int ch = ch0 + t;
    float z = zz[ch];
    float d0 = z - 2.f * dmat[t][0] + ccs[0];
    float d1 = z - 2.f * dmat[t][1] + ccs[1];
    float d2 = z - 2.f * dmat[t][2] + ccs[2];
    int lab = 0; float best = d0;
    if (d1 < best) { best = d1; lab = 1; }
    if (d2 < best) { lab = 2; }
    if (ch < CHN) labels[ch] = lab;
  }
}

// ---------------------------------------------------------------------------
// 7) K-means update: each thread owns one dim d, sweeps all 1000 points.
// ---------------------------------------------------------------------------
__global__ void __launch_bounds__(256) update_kernel(
    const u16* __restrict__ zb, const int* __restrict__ labels,
    float* __restrict__ cF, u16* __restrict__ cB, float* __restrict__ pcc) {
  __shared__ short lab[CHN];
  __shared__ int cnt[3];
  __shared__ float red[256];
  int t = threadIdx.x;
  if (t < 3) cnt[t] = 0;
  __syncthreads();
  for (int i = t; i < CHN; i += 256) {
    int l = labels[i]; lab[i] = (short)l;
    atomicAdd(&cnt[l], 1);
  }
  __syncthreads();
  int d = blockIdx.x * 256 + t;
  float a0 = 0.f, a1 = 0.f, a2 = 0.f;
  for (int n = 0; n < CHN; ++n) {
    float v = bf2f(zb[(size_t)n * KD + d]);
    int l = lab[n];
    a0 += (l == 0) ? v : 0.f;
    a1 += (l == 1) ? v : 0.f;
    a2 += (l == 2) ? v : 0.f;
  }
  float av[3] = {a0, a1, a2}, cn[3];
  #pragma unroll
  for (int k = 0; k < 3; ++k) {
    float old = cF[(size_t)k * KD + d];
    float nv  = (cnt[k] > 0) ? (av[k] / (float)cnt[k]) : old;
    cn[k] = nv;
    cF[(size_t)k * KD + d] = nv;
    cB[(size_t)k * KD + d] = f2bf(nv);
  }
  #pragma unroll
  for (int k = 0; k < 3; ++k) {
    red[t] = cn[k] * cn[k];
    __syncthreads();
    for (int s = 128; s > 0; s >>= 1) { if (t < s) red[t] += red[t + s]; __syncthreads(); }
    if (t == 0) pcc[blockIdx.x * 3 + k] = red[0];
    __syncthreads();
  }
}

// ---------------------------------------------------------------------------
// TDM: 2D tensor_load_to_lds (Wfull slab -> LDS), 2-byte elements.
// __has_builtin is evaluated per compilation pass: it is false on the host
// pass (x86), true on the gfx1250 device pass. Hard-require it only for the
// DEVICE pass; the host pass parses the manual-copy fallback.
// ---------------------------------------------------------------------------
#define HAS_TDM __has_builtin(__builtin_amdgcn_tensor_load_to_lds)
#if defined(__HIP_DEVICE_COMPILE__) && !HAS_TDM
#error "gfx1250 device pass lacks __builtin_amdgcn_tensor_load_to_lds"
#endif

#if HAS_TDM
__device__ __forceinline__ void tdm_load_2d(unsigned lds_byte_off, const void* gsrc,
                                            unsigned tile_d0, unsigned tile_d1,
                                            unsigned tensor_d0, unsigned tensor_d1,
                                            unsigned stride0_elems) {
  unsigned long long ga = (unsigned long long)(uintptr_t)gsrc;
  tdm_v4u g0;
  g0[0] = 1u;                                                   // count=1 (valid)
  g0[1] = lds_byte_off;                                         // lds_addr
  g0[2] = (unsigned)(ga & 0xFFFFFFFFull);                       // global_addr[31:0]
  g0[3] = (unsigned)((ga >> 32) & 0x1FFFFFFull) | (2u << 30);   // addr[56:32] | type=2
  tdm_v8i g1;
  g1[0] = (int)(1u << 16);                                      // data_size=1 (2B)
  g1[1] = (int)((tensor_d0 & 0xFFFFu) << 16);                   // tensor_dim0 lo
  g1[2] = (int)((tensor_d0 >> 16) | ((tensor_d1 & 0xFFFFu) << 16));
  g1[3] = (int)((tensor_d1 >> 16) | (tile_d0 << 16));           // tile_dim0
  g1[4] = (int)(tile_d1 & 0xFFFFu);                             // tile_dim1 (tile_dim2=0)
  g1[5] = (int)stride0_elems;                                   // tensor_dim0_stride lo
  g1[6] = 0; g1[7] = 0;
  tdm_v4i gz = {0, 0, 0, 0};
#if defined(__clang_major__) && (__clang_major__ >= 23)
  tdm_v8i z8 = {0, 0, 0, 0, 0, 0, 0, 0};
  __builtin_amdgcn_tensor_load_to_lds(g0, g1, gz, gz, z8, 0);
#else
  __builtin_amdgcn_tensor_load_to_lds(g0, g1, gz, gz, 0);
#endif
}
#endif // HAS_TDM

// ---------------------------------------------------------------------------
// 8) Head GEMM: out[b,p,ch] = (WMMA(Wfull[label(ch)], xn(ch)) + bias)*std + mean
//    Block = 1 channel, 128 threads (4 waves). N-tiles = 2 (b halves),
//    M-tiles = 12 (p). Wave w -> N-tile w&1, M-tiles 6*(w>>1)..+5.
//    Wfull K-slabs staged to LDS via a 2-deep TDM pipeline (TENSORcnt fenced):
//    issue slab kb+1, wait tensorcnt<=1 (slab kb done, in-order), compute kb.
// ---------------------------------------------------------------------------
__global__ void __launch_bounds__(128) head_gemm_kernel(
    const u16* __restrict__ xt, const u16* __restrict__ wfull,
    const float* __restrict__ biasf, const int* __restrict__ labels,
    const float* __restrict__ mean, const float* __restrict__ stdv,
    float* __restrict__ out) {
  __shared__ __align__(16) u16 Abuf[2][PRED * KBLK]; // 2 x 48 KB double buffer
  int ch = blockIdx.x;
  int k = labels[ch]; k = (k < 0) ? 0 : ((k > 2) ? 2 : k);
  int t = threadIdx.x;
  int lane = t & 31, w = t >> 5, hi = lane >> 4;
  int mbase = (w >> 1) * 6;
  int b = (w & 1) * 16 + (lane & 15);
  const u16* xrow = xt + ((size_t)ch * NB + b) * SEQ;          // B operand source
  const u16* wsrc = wfull + (size_t)k * PRED * SEQ;            // A operand source
  v8f acc[6];
  #pragma unroll
  for (int i = 0; i < 6; ++i) acc[i] = (v8f){0.f, 0.f, 0.f, 0.f, 0.f, 0.f, 0.f, 0.f};

  // compute K-slab kb from LDS buffer `buf`
  auto compute_slab = [&](int kb, int buf) {
    const u16* abase = &Abuf[buf][0];
    #pragma unroll
    for (int kc = 0; kc < KBLK / 32; ++kc) {
      int kg = kb * KBLK + kc * 32;
      Frag bf; // B: 16 columns = 16 batches of this channel
      bf.q[0] = *(const uv4*)(xrow + kg + hi * 8);
      bf.q[1] = *(const uv4*)(xrow + kg + 16 + hi * 8);
      #pragma unroll
      for (int tt = 0; tt < 6; ++tt) { // B reused across 6 M-tiles
        const u16* ar = abase + ((mbase + tt) * 16 + (lane & 15)) * KBLK + kc * 32;
        Frag af;
        af.q[0] = *(const uv4*)(ar + hi * 8);
        af.q[1] = *(const uv4*)(ar + 16 + hi * 8);
        acc[tt] = __builtin_amdgcn_wmma_f32_16x16x32_bf16(false, af.v, false, bf.v,
                                                          (short)0, acc[tt], false, false);
      }
    }
  };

  constexpr int NKB = SEQ / KBLK; // 8 K-slabs
#if HAS_TDM
  if (w == 0) { // one wave drives the TDM; EXEC ignored by tensor ops
    tdm_load_2d((unsigned)(uintptr_t)&Abuf[0][0], wsrc,
                KBLK, PRED, SEQ, PRED, SEQ);
  }
  for (int kb = 0; kb < NKB - 1; ++kb) {
    if (w == 0) {
      tdm_load_2d((unsigned)(uintptr_t)&Abuf[(kb + 1) & 1][0], wsrc + (kb + 1) * KBLK,
                  KBLK, PRED, SEQ, PRED, SEQ);
      __builtin_amdgcn_s_wait_tensorcnt(1); // slab kb complete (TDM is in-order)
    }
    __syncthreads();
    __builtin_prefetch(xrow + (kb + 1) * KBLK, 0, 1);
    compute_slab(kb, kb & 1);
    __syncthreads();
  }
  if (w == 0) __builtin_amdgcn_s_wait_tensorcnt(0);
  __syncthreads();
  compute_slab(NKB - 1, (NKB - 1) & 1);
#else
  // Host-pass / no-TDM fallback: cooperative global->LDS copy, single buffer.
  for (int kb = 0; kb < NKB; ++kb) {
    for (int i = t; i < PRED * KBLK / 8; i += 128) {
      int row = i >> 4, cl = (i & 15) * 8;
      *(uv4*)(&Abuf[0][0] + row * KBLK + cl) =
          *(const uv4*)(wsrc + (size_t)row * SEQ + kb * KBLK + cl);
    }
    __syncthreads();
    compute_slab(kb, 0);
    __syncthreads();
  }
#endif

  // Epilogue: bias + de-normalize, scatter to out[b, p, ch]
  float sv = stdv[(size_t)b * CHN + ch];
  float mv = mean[(size_t)b * CHN + ch];
  #pragma unroll
  for (int tt = 0; tt < 6; ++tt) {
    int p0 = (mbase + tt) * 16 + hi * 8;
    #pragma unroll
    for (int r = 0; r < 8; ++r) {
      int p = p0 + r;
      float xy = acc[tt][r] + biasf[k * PRED + p];
      out[(size_t)b * PRED * CHN + (size_t)p * CHN + ch] = xy * sv + mv;
    }
  }
}

// ---------------------------------------------------------------------------
// Launch
// ---------------------------------------------------------------------------
extern "C" void kernel_launch(void* const* d_in, const int* in_sizes, int n_in,
                              void* d_out, int out_size, void* d_ws, size_t ws_size,
                              hipStream_t stream) {
  const float* x  = (const float*)d_in[0];
  const float* Wg = (const float*)d_in[4];
  const float* bg = (const float*)d_in[5];
  const float* Wu = (const float*)d_in[6];
  const float* bu = (const float*)d_in[7];
  float* out = (float*)d_out;
  char* ws = (char*)d_ws;

  float* mean  = (float*)(ws + OFF_MEAN);
  float* rstd  = (float*)(ws + OFF_RSTD);
  float* stdv  = (float*)(ws + OFF_STD);
  float* zz    = (float*)(ws + OFF_ZZ);
  float* cF    = (float*)(ws + OFF_CF);
  u16*   cB    = (u16*)  (ws + OFF_CB);
  float* pcc   = (float*)(ws + OFF_PCC);
  int*   lab   = (int*)  (ws + OFF_LAB);
  float* biasf = (float*)(ws + OFF_BIAS);
  u16*   wf    = (u16*)  (ws + OFF_WF);
  u16*   zb    = (u16*)  (ws + OFF_ZB);
  u16*   xt    = (u16*)  (ws + OFF_XT);

  stats_kernel<<<dim3(4, 32), 256, 0, stream>>>(x, mean, rstd, stdv);
  prep_kernel<<<dim3(16, 16, 32), 256, 0, stream>>>(x, mean, rstd, xt, zb);
  zz_kernel<<<CHP, 256, 0, stream>>>(zb, zz);
  wfuse_kernel<<<(3 * PRED * SEQ + 255) / 256, 256, 0, stream>>>(Wg, Wu, bg, bu, wf, biasf);
  kinit_kernel<<<(16 * KD + 255) / 256, 256, 0, stream>>>(zb, zz, cB, cF, pcc);
  for (int it = 0; it < KMI; ++it) {
    assign_kernel<<<CHP / 16, 32, 0, stream>>>(zb, cB, zz, pcc, lab);
    update_kernel<<<KD / 256, 256, 0, stream>>>(zb, lab, cF, cB, pcc);
  }
  assign_kernel<<<CHP / 16, 32, 0, stream>>>(zb, cB, zz, pcc, lab); // final labels
  head_gemm_kernel<<<CHN, 128, 0, stream>>>(xt, wf, biasf, lab, mean, stdv, out);
}
